// PolicyNet_81458349736769
// MI455X (gfx1250) — matrix operand order repaired and checked by reading, not compiled, not run on hardware
//
#include <hip/hip_runtime.h>
#include <hip/hip_bf16.h>
#include <stdint.h>

// ---------------------------------------------------------------------------
// Problem constants (match reference)
// ---------------------------------------------------------------------------
#define NUM_GRAPHS       8
#define NODES_PER_GRAPH  10000
#define N_NODES          80000
#define N_EDGES          640000
#define F_IN             32
#define HID              128
#define AVS              128
#define N_HALF           64
#define N_ACT            500
#define N_ACT_TOT        (NUM_GRAPHS * N_ACT)   // 4000

typedef __attribute__((ext_vector_type(16))) __bf16 v16bf;
typedef __attribute__((ext_vector_type(8)))  float  v8f;

// ---------------------------------------------------------------------------
// bf16 helpers (round-to-nearest-even-ish convert, bit-exact expand)
// ---------------------------------------------------------------------------
__device__ __forceinline__ __bf16 f2bf(float f) {
  union { float f; uint32_t u; } x; x.f = f;
  uint32_t r = x.u + 0x7FFFu + ((x.u >> 16) & 1u);
  union { uint16_t u; __bf16 b; } y; y.u = (uint16_t)(r >> 16);
  return y.b;
}
__device__ __forceinline__ float bf2f(uint16_t u) {
  union { uint32_t u; float f; } x; x.u = ((uint32_t)u) << 16;
  return x.f;
}

// ---------------------------------------------------------------------------
// Small utility kernels
// ---------------------------------------------------------------------------
__global__ void zero_f32_kernel(float* __restrict__ p, int n) {
  int i = blockIdx.x * blockDim.x + threadIdx.x;
  if (i < n) p[i] = 0.0f;
}

__global__ void f32_to_bf16_kernel(const float* __restrict__ in,
                                   __bf16* __restrict__ out, int n) {
  int i = blockIdx.x * blockDim.x + threadIdx.x;
  if (i < n) out[i] = f2bf(in[i]);
}

// W is K x N row-major f32; out is N x K (column-major view of W) in bf16 so
// that a WMMA B-fragment (one column, contiguous K) is a contiguous load.
__global__ void wtrans_kernel(const float* __restrict__ W,
                              __bf16* __restrict__ out, int K, int N) {
  int i = blockIdx.x * blockDim.x + threadIdx.x;
  if (i >= K * N) return;
  int k = i / N;
  int n = i - k * N;
  out[(size_t)n * K + k] = f2bf(W[(size_t)k * N + n]);
}

// ---------------------------------------------------------------------------
// Scatter: agg[dst] += x[src]   (segment_sum over edges)
// x is bf16 [N_NODES x F]; agg is f32 [N_NODES x F]; edges = [2 x E] int32.
// One thread per (edge, 4-feature chunk): one 8B gather, four f32 atomics.
// ---------------------------------------------------------------------------
__global__ void scatter_add_kernel(const __bf16* __restrict__ x,
                                   const int* __restrict__ edges,
                                   float* __restrict__ agg,
                                   int E, int F) {
  int t = blockIdx.x * blockDim.x + threadIdx.x;
  int chunks = F >> 2;
  if (t >= E * chunks) return;
  int e = t / chunks;
  int c = (t - e * chunks) << 2;
  int src = edges[e];
  int dst = edges[E + e];
  uint2 q = *(const uint2*)(x + (size_t)src * F + c);
  float f0 = bf2f((uint16_t)(q.x & 0xFFFFu));
  float f1 = bf2f((uint16_t)(q.x >> 16));
  float f2 = bf2f((uint16_t)(q.y & 0xFFFFu));
  float f3 = bf2f((uint16_t)(q.y >> 16));
  float* a = agg + (size_t)dst * F + c;
  atomicAdd(a + 0, f0);
  atomicAdd(a + 1, f1);
  atomicAdd(a + 2, f2);
  atomicAdd(a + 3, f3);
}

// ---------------------------------------------------------------------------
// WMMA GEMM:  Y[N_NODES x 128] = A@W0 (+ X@W1) + bias
//   A      : f32 (A_F32, converted in-register) or bf16, [N_NODES x K] row-major
//   X      : bf16 [N_NODES x K] row-major (only if DUAL)
//   W0,W1  : bf16 [128 x K]  (pre-transposed: Wt[n*K+k] = W[k*128+n])
//   Y      : bf16 (OUT_BF) or f32 (!OUT_BF), [N_NODES x 128]
//
// Weights are staged in LDS once per workgroup (<=64 KB; WGP has 320 KB), so
// the inner loop's B fragments come from ds_load_b128 instead of round-tripping
// to L2 for every tile. One WAVE owns a full 16-row strip across all 8 column
// tiles: A fragments are loaded once per K-step and reused for 8 (DUAL: 16)
// v_wmma issues against 8 accumulators (CDNA5 ISA §7.12.2 layouts).
// ---------------------------------------------------------------------------
union Frag16 { v16bf v; uint4 q[2]; __bf16 e[16]; };

template <int K, bool DUAL, bool A_F32, bool OUT_BF>
__global__ void gemm_wmma_kernel(const void* __restrict__ Av,
                                 const __bf16* __restrict__ X,
                                 const __bf16* __restrict__ W0t,
                                 const __bf16* __restrict__ W1t,
                                 const float* __restrict__ bias,
                                 __bf16* __restrict__ ybf,
                                 float* __restrict__ yf32) {
  constexpr int WELEMS = 128 * K;                    // elements per weight matrix
  __shared__ __bf16 sW[(DUAL ? 2 : 1) * WELEMS];

  // --- cooperative weight stage: global -> LDS (uint4 = 8 bf16 per copy) ---
  {
    uint4* dst = (uint4*)sW;
    const uint4* s0 = (const uint4*)W0t;
    constexpr int NV = WELEMS / 8;
    for (int i = threadIdx.x; i < NV; i += 256) dst[i] = s0[i];
    if (DUAL) {
      uint4* dst1 = (uint4*)(sW + WELEMS);
      const uint4* s1 = (const uint4*)W1t;
      for (int i = threadIdx.x; i < NV; i += 256) dst1[i] = s1[i];
    }
  }
  __syncthreads();
  const __bf16* sW0 = sW;
  const __bf16* sW1 = sW + WELEMS;

  const int lane  = threadIdx.x & 31;
  const int wave  = threadIdx.x >> 5;          // 0..7
  const int strip = blockIdx.x * 8 + wave;     // 16-row strip index
  const int lrow  = lane & 15;
  const int hi    = lane >> 4;                 // 0: low K-half lanes, 1: high
  const int m     = strip * 16 + lrow;         // A row for this lane

  v8f c[8];
#pragma unroll
  for (int j = 0; j < 8; ++j) {
    float bv = bias[j * 16 + lrow];
#pragma unroll
    for (int r = 0; r < 8; ++r) c[j][r] = bv;
  }

#pragma unroll
  for (int kk = 0; kk < K; kk += 32) {
    const int ka = kk + hi * 8;    // A: lanes 0-15 hold K kk..+7 & kk+16..+23
    const int kb = kk + hi * 8;    // B: lanes 0-15 hold K kk..+7 & kk+16..+23

    Frag16 aa;
    if (A_F32) {
      const float* Af = (const float*)Av + (size_t)m * K;
      float4 f0 = *(const float4*)(Af + ka);
      float4 f1 = *(const float4*)(Af + ka + 4);
      float4 f2 = *(const float4*)(Af + ka + 16);
      float4 f3 = *(const float4*)(Af + ka + 20);
      aa.e[0]  = f2bf(f0.x); aa.e[1]  = f2bf(f0.y);
      aa.e[2]  = f2bf(f0.z); aa.e[3]  = f2bf(f0.w);
      aa.e[4]  = f2bf(f1.x); aa.e[5]  = f2bf(f1.y);
      aa.e[6]  = f2bf(f1.z); aa.e[7]  = f2bf(f1.w);
      aa.e[8]  = f2bf(f2.x); aa.e[9]  = f2bf(f2.y);
      aa.e[10] = f2bf(f2.z); aa.e[11] = f2bf(f2.w);
      aa.e[12] = f2bf(f3.x); aa.e[13] = f2bf(f3.y);
      aa.e[14] = f2bf(f3.z); aa.e[15] = f2bf(f3.w);
    } else {
      const __bf16* Ab = (const __bf16*)Av + (size_t)m * K;
      aa.q[0] = *(const uint4*)(Ab + ka);
      aa.q[1] = *(const uint4*)(Ab + ka + 16);
    }
    Frag16 ax;
    if (DUAL) {
      const __bf16* Xb = X + (size_t)m * K;
      ax.q[0] = *(const uint4*)(Xb + ka);
      ax.q[1] = *(const uint4*)(Xb + ka + 16);
    }

#pragma unroll
    for (int j = 0; j < 8; ++j) {
      const int n = j * 16 + lrow;             // output / weight column
      Frag16 b0;
      const uint4* p0 = (const uint4*)(sW0 + (size_t)n * K + kb);
      b0.q[0] = p0[0];
      b0.q[1] = p0[2];
      c[j] = __builtin_amdgcn_wmma_f32_16x16x32_bf16(false, aa.v, false, b0.v,
                                                     (short)0, c[j], false, false);
      if (DUAL) {
        Frag16 b1;
        const uint4* p1 = (const uint4*)(sW1 + (size_t)n * K + kb);
        b1.q[0] = p1[0];
        b1.q[1] = p1[2];
        c[j] = __builtin_amdgcn_wmma_f32_16x16x32_bf16(false, ax.v, false, b1.v,
                                                       (short)0, c[j], false, false);
      }
    }
  }

#pragma unroll
  for (int r = 0; r < 8; ++r) {
    const int mm = strip * 16 + r + hi * 8;    // C/D: VGPR r -> M=r (lo) / 8+r (hi)
#pragma unroll
    for (int j = 0; j < 8; ++j) {
      const int n = j * 16 + lrow;
      if (OUT_BF) ybf[(size_t)mm * 128 + n] = f2bf(c[j][r]);
      else        yf32[(size_t)mm * 128 + n] = c[j][r];
    }
  }
}

// ---------------------------------------------------------------------------
// Head: probs[b, j] = dot( dense[b, idx_s[j], 0:64], dense[b, idx_d[j], 64:128] )
// idx_s[j] = actions_flat[2*j], idx_d[j] = actions_flat[2*j+1], j in [0,4000)
// ---------------------------------------------------------------------------
__global__ void score_kernel(const float* __restrict__ dense,
                             const int* __restrict__ actions,
                             float* __restrict__ probs) {
  int idx = blockIdx.x * blockDim.x + threadIdx.x;
  if (idx >= NUM_GRAPHS * N_ACT_TOT) return;
  int b = idx / N_ACT_TOT;
  int j = idx - b * N_ACT_TOT;
  int s = actions[2 * j];
  int d = actions[2 * j + 1];
  const float4* ps = (const float4*)(dense + (size_t)(b * NODES_PER_GRAPH + s) * AVS);
  const float4* pd = (const float4*)(dense + (size_t)(b * NODES_PER_GRAPH + d) * AVS + N_HALF);
  float acc = 0.0f;
#pragma unroll
  for (int i = 0; i < N_HALF / 4; ++i) {
    float4 a = ps[i], bb = pd[i];
    acc += a.x * bb.x + a.y * bb.y + a.z * bb.z + a.w * bb.w;
  }
  probs[idx] = acc;
}

// Softmax over the 4000-long action axis, one block per graph.
__global__ void softmax_kernel(const float* __restrict__ probs,
                               float* __restrict__ out) {
  __shared__ float red[256];
  const int b = blockIdx.x;
  const int t = threadIdx.x;
  const float* row = probs + (size_t)b * N_ACT_TOT;
  float* orow = out + (size_t)b * N_ACT_TOT;

  float mx = -3.0e38f;
  for (int j = t; j < N_ACT_TOT; j += 256) mx = fmaxf(mx, row[j]);
  red[t] = mx;
  __syncthreads();
  for (int s = 128; s > 0; s >>= 1) {
    if (t < s) red[t] = fmaxf(red[t], red[t + s]);
    __syncthreads();
  }
  mx = red[0];
  __syncthreads();

  float sum = 0.0f;
  for (int j = t; j < N_ACT_TOT; j += 256) sum += __expf(row[j] - mx);
  red[t] = sum;
  __syncthreads();
  for (int s = 128; s > 0; s >>= 1) {
    if (t < s) red[t] += red[t + s];
    __syncthreads();
  }
  float inv = 1.0f / red[0];
  for (int j = t; j < N_ACT_TOT; j += 256) orow[j] = __expf(row[j] - mx) * inv;
}

// ---------------------------------------------------------------------------
// Launch
// ---------------------------------------------------------------------------
extern "C" void kernel_launch(void* const* d_in, const int* in_sizes, int n_in,
                              void* d_out, int out_size, void* d_ws, size_t ws_size,
                              hipStream_t stream) {
  (void)in_sizes; (void)n_in; (void)out_size; (void)ws_size;

  const int*   actions = (const int*)d_in[0];
  const float* obs     = (const float*)d_in[1];
  const int*   eic     = (const int*)d_in[2];
  const int*   eid     = (const int*)d_in[3];
  const int*   eit     = (const int*)d_in[4];
  // d_in[5] = batch (unused: graphs are contiguous blocks of 10000 nodes)
  const float* W_rel1  = (const float*)d_in[6];
  const float* W_root1 = (const float*)d_in[7];
  const float* b1      = (const float*)d_in[8];
  const float* W_rel2  = (const float*)d_in[9];
  const float* W_root2 = (const float*)d_in[10];
  const float* b2      = (const float*)d_in[11];
  const float* W_rel3  = (const float*)d_in[12];
  const float* W_root3 = (const float*)d_in[13];
  const float* b3      = (const float*)d_in[14];
  const float* W_l0    = (const float*)d_in[15];
  const float* b_l0    = (const float*)d_in[16];
  const float* W_l1    = (const float*)d_in[17];
  const float* b_l1    = (const float*)d_in[18];
  const float* W_out   = (const float*)d_in[19];
  const float* b_out   = (const float*)d_in[20];

  // Workspace layout (bytes); total ~82.3 MB
  char* ws = (char*)d_ws;
  __bf16* xbuf0 = (__bf16*)(ws + 0);           // 80000*128 bf16 = 20,480,000 B
  __bf16* xbuf1 = (__bf16*)(ws + 20480000);    // 20,480,000 B
  float*  aggf  = (float*)(ws + 40960000);     // 80000*128 f32 = 40,960,000 B
  __bf16* wbuf  = (__bf16*)(ws + 81920000);    // 122,880 elems = 245,760 B
  float*  probs = (float*)(ws + 82165760);     // 32,000 f32 = 128,000 B
  float*  dense = aggf;                        // reuse aggf for final f32 acts

  __bf16* Wrel1t  = wbuf + 0;        // 128 x 32
  __bf16* Wroot1t = wbuf + 4096;     // 128 x 32
  __bf16* Wrel2t  = wbuf + 8192;     // 128 x 128
  __bf16* Wroot2t = wbuf + 24576;
  __bf16* Wrel3t  = wbuf + 40960;
  __bf16* Wroot3t = wbuf + 57344;
  __bf16* Wl0t    = wbuf + 73728;
  __bf16* Wl1t    = wbuf + 90112;
  __bf16* Woutt   = wbuf + 106496;

  auto g1 = [](int n) { return dim3((unsigned)((n + 255) / 256)); };
  const dim3 blk(256);
  const dim3 gemm_grid(N_NODES / 16 / 8);      // 625 blocks x 8 waves = 5000 strips

  // --- weights -> bf16 transposed ---
  wtrans_kernel<<<g1(32 * 128), blk, 0, stream>>>(W_rel1,  Wrel1t,  32,  128);
  wtrans_kernel<<<g1(32 * 128), blk, 0, stream>>>(W_root1, Wroot1t, 32,  128);
  wtrans_kernel<<<g1(128 * 128), blk, 0, stream>>>(W_rel2,  Wrel2t,  128, 128);
  wtrans_kernel<<<g1(128 * 128), blk, 0, stream>>>(W_root2, Wroot2t, 128, 128);
  wtrans_kernel<<<g1(128 * 128), blk, 0, stream>>>(W_rel3,  Wrel3t,  128, 128);
  wtrans_kernel<<<g1(128 * 128), blk, 0, stream>>>(W_root3, Wroot3t, 128, 128);
  wtrans_kernel<<<g1(128 * 128), blk, 0, stream>>>(W_l0,    Wl0t,    128, 128);
  wtrans_kernel<<<g1(128 * 128), blk, 0, stream>>>(W_l1,    Wl1t,    128, 128);
  wtrans_kernel<<<g1(128 * 128), blk, 0, stream>>>(W_out,   Woutt,   128, 128);

  // --- obs -> bf16 (N_NODES x 32 in xbuf0) ---
  f32_to_bf16_kernel<<<g1(N_NODES * F_IN), blk, 0, stream>>>(obs, xbuf0, N_NODES * F_IN);

  // --- layer 1 (edges: eit, F = 32); agg stays f32, converted inside GEMM ---
  zero_f32_kernel<<<g1(N_NODES * F_IN), blk, 0, stream>>>(aggf, N_NODES * F_IN);
  scatter_add_kernel<<<g1(N_EDGES * (F_IN / 4)), blk, 0, stream>>>(xbuf0, eit, aggf, N_EDGES, F_IN);
  gemm_wmma_kernel<32, true, true, true><<<gemm_grid, blk, 0, stream>>>(
      aggf, xbuf0, Wrel1t, Wroot1t, b1, xbuf1, nullptr);

  // --- layer 2 (edges: eic, F = 128) ---
  zero_f32_kernel<<<g1(N_NODES * HID), blk, 0, stream>>>(aggf, N_NODES * HID);
  scatter_add_kernel<<<g1(N_EDGES * (HID / 4)), blk, 0, stream>>>(xbuf1, eic, aggf, N_EDGES, HID);
  gemm_wmma_kernel<128, true, true, true><<<gemm_grid, blk, 0, stream>>>(
      aggf, xbuf1, Wrel2t, Wroot2t, b2, xbuf0, nullptr);

  // --- layer 3 (edges: eid, F = 128) ---
  zero_f32_kernel<<<g1(N_NODES * HID), blk, 0, stream>>>(aggf, N_NODES * HID);
  scatter_add_kernel<<<g1(N_EDGES * (HID / 4)), blk, 0, stream>>>(xbuf0, eid, aggf, N_EDGES, HID);
  gemm_wmma_kernel<128, true, true, true><<<gemm_grid, blk, 0, stream>>>(
      aggf, xbuf0, Wrel3t, Wroot3t, b3, xbuf1, nullptr);

  // --- dense layers l0, l1, out ---
  gemm_wmma_kernel<128, false, false, true><<<gemm_grid, blk, 0, stream>>>(
      xbuf1, nullptr, Wl0t, nullptr, b_l0, xbuf0, nullptr);
  gemm_wmma_kernel<128, false, false, true><<<gemm_grid, blk, 0, stream>>>(
      xbuf0, nullptr, Wl1t, nullptr, b_l1, xbuf1, nullptr);
  gemm_wmma_kernel<128, false, false, false><<<gemm_grid, blk, 0, stream>>>(
      xbuf1, nullptr, Woutt, nullptr, b_out, nullptr, dense);

  // --- head: action-pair dots + per-graph softmax ---
  score_kernel<<<g1(NUM_GRAPHS * N_ACT_TOT), blk, 0, stream>>>(dense, actions, probs);
  softmax_kernel<<<NUM_GRAPHS, blk, 0, stream>>>(probs, (float*)d_out);
}
